// AdvancedHierarchicalS2Model_31653908971573
// MI455X (gfx1250) — matrix operand order repaired and checked by reading, not compiled, not run on hardware
//
#include <hip/hip_runtime.h>
#include <hip/hip_bf16.h>
#include <math.h>

// ---------------------------------------------------------------------------
// Model constants
// ---------------------------------------------------------------------------
#define Bc    32
#define Tc    100
#define Dm    512
#define Hh    8
#define HD    64
#define VX    40000
#define V11   512
#define V13   4096
#define V14   16384
#define COMBP 704            // 682 padded to multiple of 32
#define COMBR 682
#define GHd   256
#define HIDd  256

typedef __attribute__((ext_vector_type(16))) _Float16 v16h;
typedef __attribute__((ext_vector_type(8)))  float    v8f;

// gfx1250 async global->LDS path (guarded; falls back to direct loads)
#if defined(__AMDGCN__) && __has_builtin(__builtin_amdgcn_global_load_async_to_lds_b128) && __has_builtin(__builtin_amdgcn_s_wait_asynccnt)
#define USE_ASYNC_LDS 1
#else
#define USE_ASYNC_LDS 0
#endif

#if USE_ASYNC_LDS
// builtin prototype (from compiler diagnostic): param is pointer to
// `int __attribute__((vector_size(16)))` in AS(1) (global) / AS(3) (LDS)
typedef int v4i_ __attribute__((vector_size(4 * sizeof(int))));
typedef __attribute__((address_space(1))) v4i_* gp128;
typedef __attribute__((address_space(3))) v4i_* lp128;
// stage 16 consecutive floats (one lane's share of a 16x32 A tile) into LDS
__device__ __forceinline__ void async_stage16(const float* g, float* l)
{
    #pragma unroll
    for (int i = 0; i < 4; ++i)
        __builtin_amdgcn_global_load_async_to_lds_b128(
            (gp128)(g + 4 * i), (lp128)(l + 4 * i), 0, 0);
}
#endif

// ---------------------------------------------------------------------------
// A-fragment (16x32 f16) from an f32 row pointer:
// lanes 0-15 hold K {0..7,16..23}, lanes 16-31 hold K {8..15,24..31}
// ---------------------------------------------------------------------------
__device__ __forceinline__ v16h afrag_from(const float* p /* row + (hi?8:0) */)
{
    v16h a;
    #pragma unroll
    for (int i = 0; i < 8; ++i) a[i]     = (_Float16)p[i];
    #pragma unroll
    for (int i = 0; i < 8; ++i) a[8 + i] = (_Float16)p[16 + i];
    return a;
}
__device__ __forceinline__ v16h bfrag_from(const float* p /* Wrow + (hi?8:0) */)
{
    v16h b;
    #pragma unroll
    for (int i = 0; i < 8; ++i) b[i]     = (_Float16)p[i];
    #pragma unroll
    for (int i = 0; i < 8; ++i) b[8 + i] = (_Float16)p[16 + i];
    return b;
}

// ---------------------------------------------------------------------------
// WMMA GEMM:  C[M,N] (+)= A[M,K] * W[N,K]^T
// One wave computes a 16x64 macro-tile: 4 v_wmma per K-chunk sharing one
// A-fragment (4x A-reuse).  grid = (N/64, M/16, Ksplit).
// A tile is double-buffer staged into LDS with async loads when available.
// ---------------------------------------------------------------------------
__global__ __launch_bounds__(32)
void wmma_gemm(const float* __restrict__ A, int lda,
               const float* __restrict__ W, int ldw,
               float* __restrict__ C, int ldc,
               int K, int accumulate)
{
    const int lane = threadIdx.x;
    const int n0   = blockIdx.x * 64;
    const int m0   = blockIdx.y * 16;

    const int kchunks = K >> 5;                 // K multiple of 32 by construction
    const int per     = (kchunks + (int)gridDim.z - 1) / (int)gridDim.z;
    int c0 = (int)blockIdx.z * per;
    int c1 = c0 + per; if (c1 > kchunks) c1 = kchunks;

    const int r  = lane & 15;
    const int hi = lane >> 4;

    const float* Arow = A + (long)(m0 + r) * lda;
    const float* Wr0 = W + (long)(n0 + r)      * ldw;
    const float* Wr1 = W + (long)(n0 + 16 + r) * ldw;
    const float* Wr2 = W + (long)(n0 + 32 + r) * ldw;
    const float* Wr3 = W + (long)(n0 + 48 + r) * ldw;

    v8f acc0 = {}, acc1 = {}, acc2 = {}, acc3 = {};

#if USE_ASYNC_LDS
    __shared__ float abuf[2][16 * 32];
    if (c0 < c1)
        async_stage16(Arow + (c0 << 5) + hi * 16, &abuf[c0 & 1][r * 32 + hi * 16]);
#endif

    for (int c = c0; c < c1; ++c) {
        const int k0 = c << 5;
        v16h a;
#if USE_ASYNC_LDS
        __builtin_amdgcn_s_wait_asynccnt(0);        // current A tile resident in LDS
        a = afrag_from(&abuf[c & 1][r * 32 + (hi ? 8 : 0)]);
        if (c + 1 < c1)                              // prefetch next A tile
            async_stage16(Arow + ((c + 1) << 5) + hi * 16,
                          &abuf[(c + 1) & 1][r * 32 + hi * 16]);
#else
        a = afrag_from(Arow + k0 + (hi ? 8 : 0));
#endif
        __builtin_prefetch(Wr0 + k0 + 64, 0, 1);     // global_prefetch_b8 weight stream
        const int ko = k0 + (hi ? 8 : 0);
        v16h b0 = bfrag_from(Wr0 + ko);
        v16h b1 = bfrag_from(Wr1 + ko);
        v16h b2 = bfrag_from(Wr2 + ko);
        v16h b3 = bfrag_from(Wr3 + ko);
        acc0 = __builtin_amdgcn_wmma_f32_16x16x32_f16(false, a, false, b0, (short)0, acc0, false, false);
        acc1 = __builtin_amdgcn_wmma_f32_16x16x32_f16(false, a, false, b1, (short)0, acc1, false, false);
        acc2 = __builtin_amdgcn_wmma_f32_16x16x32_f16(false, a, false, b2, (short)0, acc2, false, false);
        acc3 = __builtin_amdgcn_wmma_f32_16x16x32_f16(false, a, false, b3, (short)0, acc3, false, false);
    }

    // C/D layout: lanes 0-15 -> (row v, col lane); lanes 16-31 -> (row 8+v, col lane-16)
    float* Cp = C + (long)(m0 + (hi << 3)) * ldc + n0 + r;
    if (accumulate) {
        #pragma unroll
        for (int v = 0; v < 8; ++v) {
            atomicAdd(Cp + (long)v * ldc,      acc0[v]);
            atomicAdd(Cp + (long)v * ldc + 16, acc1[v]);
            atomicAdd(Cp + (long)v * ldc + 32, acc2[v]);
            atomicAdd(Cp + (long)v * ldc + 48, acc3[v]);
        }
    } else {
        #pragma unroll
        for (int v = 0; v < 8; ++v) {
            Cp[(long)v * ldc]      = acc0[v];
            Cp[(long)v * ldc + 16] = acc1[v];
            Cp[(long)v * ldc + 32] = acc2[v];
            Cp[(long)v * ldc + 48] = acc3[v];
        }
    }
}

// ---------------------------------------------------------------------------
// Small helper kernels
// ---------------------------------------------------------------------------
__global__ void fill_kernel(float* p, long n)
{
    long i = (long)blockIdx.x * 256 + threadIdx.x;
    if (i < n) p[i] = 0.f;
}

__global__ void pad_copy_kernel(const float* __restrict__ src, float* __restrict__ dst,
                                int rows, int scols, int dcols)
{
    long i = (long)blockIdx.x * 256 + threadIdx.x;
    if (i >= (long)rows * dcols) return;
    int r = (int)(i / dcols), c = (int)(i % dcols);
    dst[i] = (c < scols) ? src[(long)r * scols + c] : 0.f;
}

__global__ void bias_act_kernel(float* __restrict__ C, int ldc,
                                const float* __restrict__ bias,
                                int rows, int cols, int relu)
{
    long i = (long)blockIdx.x * 256 + threadIdx.x;
    if (i >= (long)rows * cols) return;
    int r = (int)(i / cols), c = (int)(i % cols);
    float v = C[(long)r * ldc + c] + bias[c];
    if (relu) v = fmaxf(v, 0.f);
    C[(long)r * ldc + c] = v;
}

// freq / recency histograms + normalization (per batch row)
__global__ __launch_bounds__(256)
void stats_kernel(const int* __restrict__ X, const float* __restrict__ decay_p,
                  float* __restrict__ fd, float* __restrict__ rd)
{
    const int b = blockIdx.x, tid = threadIdx.x;
    float* f = fd + (long)b * VX;
    float* r = rd + (long)b * VX;
    for (int i = tid; i < VX; i += 256) { f[i] = 0.f; r[i] = 0.f; }
    __syncthreads();
    const float decay = decay_p[0];
    for (int t = tid; t < Tc; t += 256) {
        int x = X[b * Tc + t];
        atomicAdd(&f[x], 1.f);
        atomicAdd(&r[x], __powf(decay, (float)(Tc - 1 - t)));
    }
    __threadfence_block();
    __syncthreads();
    __shared__ float sf[256], sr[256];
    float a = 0.f, c = 0.f;
    for (int i = tid; i < VX; i += 256) { a += f[i]; c += r[i]; }
    sf[tid] = a; sr[tid] = c; __syncthreads();
    for (int s = 128; s; s >>= 1) {
        if (tid < s) { sf[tid] += sf[tid + s]; sr[tid] += sr[tid + s]; }
        __syncthreads();
    }
    const float fi = 1.f / (sf[0] + 1e-8f), ri = 1.f / (sr[0] + 1e-8f);
    for (int i = tid; i < VX; i += 256) { f[i] *= fi; r[i] *= ri; }
}

// gather embeddings -> padded concat row (B*T, 704)
__global__ __launch_bounds__(256)
void loc_stage_kernel(const int* __restrict__ l11, const int* __restrict__ l13,
                      const int* __restrict__ l14, const int* __restrict__ X,
                      const float* __restrict__ e11, const float* __restrict__ e13,
                      const float* __restrict__ e14, const float* __restrict__ eX,
                      float* __restrict__ loc)
{
    const int row = blockIdx.x, tid = threadIdx.x;
    const int i11 = l11[row], i13 = l13[row], i14 = l14[row], iX = X[row];
    float* o = loc + (long)row * COMBP;
    for (int c = tid; c < COMBP; c += 256) {
        float v;
        if      (c < 128) v = e11[(long)i11 * 128 + c];
        else if (c < 256) v = e13[(long)i13 * 128 + (c - 128)];
        else if (c < 426) v = e14[(long)i14 * 170 + (c - 256)];
        else if (c < 682) v = eX [(long)iX  * 256 + (c - 426)];
        else              v = 0.f;
        o[c] = v;
    }
}

// xproj += loc_b + pe[t]
__global__ void xproj_epi_kernel(float* __restrict__ Xp, const float* __restrict__ bias,
                                 const float* __restrict__ pe)
{
    long i = (long)blockIdx.x * 256 + threadIdx.x;
    if (i >= (long)Bc * Tc * Dm) return;
    int d = (int)(i & (Dm - 1));
    int t = (int)((i >> 9) % Tc);
    Xp[i] += bias[d] + pe[t * Dm + d];
}

// ---------------------------------------------------------------------------
// GRU recurrence: one workgroup (8 waves), h (32x256) in LDS.
// Each step: gh(32x768) = h @ Whh^T -> 2x12 macro-tiles of 16x64
// (4 v_wmma share one LDS A-fragment), then VALU gate math updates h.
// ---------------------------------------------------------------------------
__global__ __launch_bounds__(256)
void gru_kernel(const float* __restrict__ gi,   // (B*T, 768) = x@Wih^T + bih
                const float* __restrict__ Whh,  // (768, 256)
                const float* __restrict__ bhh,  // (768)
                float* __restrict__ ghbuf,      // (32, 768) scratch
                float* __restrict__ gru_h)      // (B*T, 256)
{
    __shared__ float h[Bc * GHd];               // 32 KB
    const int tid  = threadIdx.x;
    const int lane = tid & 31;
    const int wave = tid >> 5;

    for (int i = tid; i < Bc * GHd; i += 256) h[i] = 0.f;
    __syncthreads();

    for (int t = 0; t < Tc; ++t) {
        // gh = h @ Whh^T : M=32, N=768, K=256 -> (2 M) x (12 N-macro) = 24 macro-tiles
        for (int mt = wave; mt < 24; mt += 8) {
            const int m0 = (mt & 1) * 16;
            const int n0 = (mt >> 1) * 64;
            const int r  = lane & 15;
            const int hi = lane >> 4;
            const float* hrow = &h[(m0 + r) * GHd];
            const float* wr0 = Whh + (long)(n0 + r)      * GHd;
            const float* wr1 = Whh + (long)(n0 + 16 + r) * GHd;
            const float* wr2 = Whh + (long)(n0 + 32 + r) * GHd;
            const float* wr3 = Whh + (long)(n0 + 48 + r) * GHd;
            v8f a0 = {}, a1 = {}, a2 = {}, a3 = {};
            #pragma unroll
            for (int k0 = 0; k0 < GHd; k0 += 32) {
                v16h a = afrag_from(hrow + k0 + (hi ? 8 : 0));
                const int ko = k0 + (hi ? 8 : 0);
                v16h b0 = bfrag_from(wr0 + ko);
                v16h b1 = bfrag_from(wr1 + ko);
                v16h b2 = bfrag_from(wr2 + ko);
                v16h b3 = bfrag_from(wr3 + ko);
                a0 = __builtin_amdgcn_wmma_f32_16x16x32_f16(false, a, false, b0, (short)0, a0, false, false);
                a1 = __builtin_amdgcn_wmma_f32_16x16x32_f16(false, a, false, b1, (short)0, a1, false, false);
                a2 = __builtin_amdgcn_wmma_f32_16x16x32_f16(false, a, false, b2, (short)0, a2, false, false);
                a3 = __builtin_amdgcn_wmma_f32_16x16x32_f16(false, a, false, b3, (short)0, a3, false, false);
            }
            float* gp = ghbuf + (long)(m0 + (hi << 3)) * 768 + n0 + r;
            #pragma unroll
            for (int v = 0; v < 8; ++v) {
                gp[(long)v * 768]      = a0[v];
                gp[(long)v * 768 + 16] = a1[v];
                gp[(long)v * 768 + 32] = a2[v];
                gp[(long)v * 768 + 48] = a3[v];
            }
        }
        __threadfence_block();
        __syncthreads();

        // gate math
        for (int i = tid; i < Bc * GHd; i += 256) {
            const int b = i >> 8, n = i & 255;
            const float* g  = gi + (long)(b * Tc + t) * 768;
            const float* gh = ghbuf + (long)b * 768;
            const float hr = gh[n]       + bhh[n];
            const float hz = gh[256 + n] + bhh[256 + n];
            const float hn = gh[512 + n] + bhh[512 + n];
            const float rg = 1.f / (1.f + __expf(-(g[n]       + hr)));
            const float zg = 1.f / (1.f + __expf(-(g[256 + n] + hz)));
            const float nn = tanhf(g[512 + n] + rg * hn);
            const float hv = (1.f - zg) * nn + zg * h[i];
            h[i] = hv;
            gru_h[(long)(b * Tc + t) * GHd + n] = hv;
        }
        __syncthreads();
    }
}

// layernorm over 512 per row, with affine
__global__ __launch_bounds__(256)
void ln_kernel(float* __restrict__ X, const float* __restrict__ g, const float* __restrict__ bta)
{
    float* x = X + (long)blockIdx.x * Dm;
    const int tid = threadIdx.x;
    __shared__ float s[256];
    const float v0 = x[tid], v1 = x[tid + 256];
    s[tid] = v0 + v1; __syncthreads();
    for (int st = 128; st; st >>= 1) { if (tid < st) s[tid] += s[tid + st]; __syncthreads(); }
    const float mu = s[0] * (1.f / Dm);
    __syncthreads();
    const float d0 = v0 - mu, d1 = v1 - mu;
    s[tid] = d0 * d0 + d1 * d1; __syncthreads();
    for (int st = 128; st; st >>= 1) { if (tid < st) s[tid] += s[tid + st]; __syncthreads(); }
    const float inv = rsqrtf(s[0] * (1.f / Dm) + 1e-5f);
    x[tid]       = d0 * inv * g[tid]       + bta[tid];
    x[tid + 256] = d1 * inv * g[tid + 256] + bta[tid + 256];
}

// attention per (b,h): scores -> softmax -> ctx; also emits attn weights
__global__ __launch_bounds__(128)
void attn_kernel(const float* __restrict__ Q, const float* __restrict__ Kb,
                 const float* __restrict__ Vb, float* __restrict__ attn,
                 float* __restrict__ ctx)
{
    const int bh = blockIdx.x, b = bh >> 3, h = bh & 7;
    const int tid = threadIdx.x;
    __shared__ float sc[128], red[128];
    float s = -1e30f;
    if (tid < Tc) {
        const float* q = Q  + (long)b * Dm + h * HD;
        const float* k = Kb + (long)(b * Tc + tid) * Dm + h * HD;
        float d = 0.f;
        #pragma unroll 8
        for (int i = 0; i < HD; ++i) d += q[i] * k[i];
        s = d * 0.125f;                         // 1/sqrt(64)
    }
    red[tid] = s; __syncthreads();
    for (int st = 64; st; st >>= 1) { if (tid < st) red[tid] = fmaxf(red[tid], red[tid + st]); __syncthreads(); }
    const float mx = red[0]; __syncthreads();
    const float e = (tid < Tc) ? __expf(s - mx) : 0.f;
    red[tid] = e; __syncthreads();
    for (int st = 64; st; st >>= 1) { if (tid < st) red[tid] += red[tid + st]; __syncthreads(); }
    const float w = e / red[0];
    sc[tid] = w;
    if (tid < Tc) attn[(long)(b * Hh + h) * Tc + tid] = w;
    __syncthreads();
    if (tid < HD) {
        float acc = 0.f;
        for (int t = 0; t < Tc; ++t)
            acc += sc[t] * Vb[(long)(b * Tc + t) * Dm + h * HD + tid];
        ctx[(long)b * Dm + h * HD + tid] = acc;
    }
}

// combined = final_state + user_context + attended + pad(user_emb)
__global__ void combined_kernel(const float* __restrict__ gln, const float* __restrict__ uc,
                                const float* __restrict__ att, const float* __restrict__ uemb,
                                const int* __restrict__ useq, float* __restrict__ comb)
{
    long i = (long)blockIdx.x * 256 + threadIdx.x;
    if (i >= (long)Bc * Dm) return;
    const int b = (int)(i >> 9), d = (int)(i & 511);
    const float fs = gln[(long)(b * Tc + Tc - 1) * Dm + d];
    const float uf = (d < 256) ? uemb[(long)useq[b * Tc] * 256 + d] : 0.f;
    comb[i] = fs + uc[i] + att[i] + uf;
}

// copy_prob = sigmoid(g1 . g2_W + g2_b)
__global__ __launch_bounds__(64)
void g2_kernel(const float* __restrict__ g1, const float* __restrict__ W,
               const float* __restrict__ bb, float* __restrict__ cp)
{
    const int b = blockIdx.x, tid = threadIdx.x;
    __shared__ float s[64];
    float a = 0.f;
    for (int i = tid; i < Dm; i += 64) a += g1[(long)b * Dm + i] * W[i];
    s[tid] = a; __syncthreads();
    for (int st = 32; st; st >>= 1) { if (tid < st) s[tid] += s[tid + st]; __syncthreads(); }
    if (tid == 0) cp[b] = 1.f / (1.f + __expf(-(s[0] + bb[0])));
}

// copy_dist = norm(scatter(attn.mean(1)) + 0.3*freq + 0.2*rec)
__global__ __launch_bounds__(256)
void copydist_kernel(const int* __restrict__ X, const float* __restrict__ attn,
                     const float* __restrict__ fd, const float* __restrict__ rd,
                     float* __restrict__ cd)
{
    const int b = blockIdx.x, tid = threadIdx.x;
    float* c = cd + (long)b * VX;
    const float* f = fd + (long)b * VX;
    const float* r = rd + (long)b * VX;
    for (int i = tid; i < VX; i += 256) c[i] = 0.3f * f[i] + 0.2f * r[i];
    __threadfence_block(); __syncthreads();
    for (int t = tid; t < Tc; t += 256) {
        float w = 0.f;
        #pragma unroll
        for (int h = 0; h < Hh; ++h) w += attn[(long)(b * Hh + h) * Tc + t];
        atomicAdd(&c[X[b * Tc + t]], w * 0.125f);
    }
    __threadfence_block(); __syncthreads();
    __shared__ float s[256];
    float a = 0.f;
    for (int i = tid; i < VX; i += 256) a += c[i];
    s[tid] = a; __syncthreads();
    for (int st = 128; st; st >>= 1) { if (tid < st) s[tid] += s[tid + st]; __syncthreads(); }
    const float inv = 1.f / (s[0] + 1e-8f);
    for (int i = tid; i < VX; i += 256) c[i] *= inv;
}

// logits_X = log(cp*copy_dist + (1-cp)*softmax(gen) + 1e-10)
__global__ __launch_bounds__(256)
void final_kernel(const float* __restrict__ gen, const float* __restrict__ cd,
                  const float* __restrict__ cp, float* __restrict__ out)
{
    const int b = blockIdx.x, tid = threadIdx.x;
    const float* gl = gen + (long)b * VX;
    const float* c  = cd  + (long)b * VX;
    float* o = out + (long)b * VX;
    __shared__ float s[256];
    float mx = -1e30f;
    for (int i = tid; i < VX; i += 256) mx = fmaxf(mx, gl[i]);
    s[tid] = mx; __syncthreads();
    for (int st = 128; st; st >>= 1) { if (tid < st) s[tid] = fmaxf(s[tid], s[tid + st]); __syncthreads(); }
    mx = s[0]; __syncthreads();
    float sum = 0.f;
    for (int i = tid; i < VX; i += 256) sum += __expf(gl[i] - mx);
    s[tid] = sum; __syncthreads();
    for (int st = 128; st; st >>= 1) { if (tid < st) s[tid] += s[tid + st]; __syncthreads(); }
    const float inv = 1.f / s[0];
    const float p = cp[b];
    for (int i = tid; i < VX; i += 256)
        o[i] = __logf(p * c[i] + (1.f - p) * __expf(gl[i] - mx) * inv + 1e-10f);
}

// ---------------------------------------------------------------------------
// Host orchestration
// ---------------------------------------------------------------------------
extern "C" void kernel_launch(void* const* d_in, const int* in_sizes, int n_in,
                              void* d_out, int out_size, void* d_ws, size_t ws_size,
                              hipStream_t stream)
{
    (void)in_sizes; (void)n_in; (void)out_size; (void)ws_size;
    const int* l11  = (const int*)d_in[0];
    const int* l13  = (const int*)d_in[1];
    const int* l14  = (const int*)d_in[2];
    const int* Xs   = (const int*)d_in[3];
    const int* usr  = (const int*)d_in[4];
    const float* user_emb = (const float*)d_in[5];
    const float* e11 = (const float*)d_in[6];
    const float* e13 = (const float*)d_in[7];
    const float* e14 = (const float*)d_in[8];
    const float* eX  = (const float*)d_in[9];
    const float* loc_W = (const float*)d_in[10];
    const float* loc_b = (const float*)d_in[11];
    const float* freq_W = (const float*)d_in[12];
    const float* freq_b = (const float*)d_in[13];
    const float* rec_W  = (const float*)d_in[14];
    const float* rec_b  = (const float*)d_in[15];
    const float* comb_W = (const float*)d_in[16];
    const float* comb_b = (const float*)d_in[17];
    const float* decay  = (const float*)d_in[18];
    const float* gru_Wih = (const float*)d_in[19];
    const float* gru_Whh = (const float*)d_in[20];
    const float* gru_bih = (const float*)d_in[21];
    const float* gru_bhh = (const float*)d_in[22];
    const float* gp_W = (const float*)d_in[23];
    const float* gp_b = (const float*)d_in[24];
    const float* ln_g = (const float*)d_in[25];
    const float* ln_b = (const float*)d_in[26];
    const float* q_W = (const float*)d_in[27];
    const float* q_b = (const float*)d_in[28];
    const float* k_W = (const float*)d_in[29];
    const float* k_b = (const float*)d_in[30];
    const float* v_W = (const float*)d_in[31];
    const float* v_b = (const float*)d_in[32];
    const float* o_W = (const float*)d_in[33];
    const float* o_b = (const float*)d_in[34];
    const float* g1_W = (const float*)d_in[35];
    const float* g1_b = (const float*)d_in[36];
    const float* g2_W = (const float*)d_in[37];
    const float* g2_b = (const float*)d_in[38];
    const float* fc_W = (const float*)d_in[39];
    const float* fc_b = (const float*)d_in[40];
    const float* c11_W = (const float*)d_in[41];
    const float* c11_b = (const float*)d_in[42];
    const float* c13_W = (const float*)d_in[43];
    const float* c13_b = (const float*)d_in[44];
    const float* c14_W = (const float*)d_in[45];
    const float* c14_b = (const float*)d_in[46];
    const float* cX_W = (const float*)d_in[47];
    const float* cX_b = (const float*)d_in[48];
    const float* pe   = (const float*)d_in[49];

    float* out11 = (float*)d_out;                  // 32 x 512
    float* out13 = out11 + (long)Bc * V11;         // 32 x 4096
    float* out14 = out13 + (long)Bc * V13;         // 32 x 16384
    float* outX  = out14 + (long)Bc * V14;         // 32 x 40000

    float* w = (float*)d_ws;
    size_t off = 0;
    auto alloc = [&](size_t n) { float* p = w + off; off += n; return p; };
    float* fd      = alloc((size_t)Bc * VX);
    float* rd      = alloc((size_t)Bc * VX);
    float* fr_cat  = alloc((size_t)Bc * 1024);
    float* uc      = alloc((size_t)Bc * Dm);
    float* locbuf  = alloc((size_t)Bc * Tc * COMBP);
    float* locWp   = alloc((size_t)Dm * COMBP);
    float* xproj   = alloc((size_t)Bc * Tc * Dm);
    float* gi      = alloc((size_t)Bc * Tc * 768);
    float* ghbuf   = alloc((size_t)Bc * 768);
    float* gruh    = alloc((size_t)Bc * Tc * GHd);
    float* gln     = alloc((size_t)Bc * Tc * Dm);
    float* Kb      = alloc((size_t)Bc * Tc * Dm);
    float* Vb      = alloc((size_t)Bc * Tc * Dm);
    float* Qb      = alloc((size_t)Bc * Dm);
    float* attnw   = alloc((size_t)Bc * Hh * Tc);
    float* ctxb    = alloc((size_t)Bc * Dm);
    float* attd    = alloc((size_t)Bc * Dm);
    float* comb    = alloc((size_t)Bc * Dm);
    float* hid     = alloc((size_t)Bc * HIDd);
    float* g1      = alloc((size_t)Bc * Dm);
    float* cp      = alloc((size_t)Bc);
    float* gen     = alloc((size_t)Bc * VX);
    float* cd      = alloc((size_t)Bc * VX);

    // N must be a multiple of 64 (16x64 macro-tiles) — true for all uses here.
    auto gemm = [&](const float* A, int lda, const float* W, int ldw,
                    float* C, int ldc, int M, int N, int K,
                    int ksplit, int acc) {
        dim3 g(N / 64, M / 16, ksplit);
        wmma_gemm<<<g, dim3(32), 0, stream>>>(A, lda, W, ldw, C, ldc, K, acc);
    };
    auto bias = [&](float* C, int ldc, const float* b, int rows, int cols, int relu) {
        long n = (long)rows * cols;
        bias_act_kernel<<<dim3((unsigned)((n + 255) / 256)), dim3(256), 0, stream>>>(
            C, ldc, b, rows, cols, relu);
    };
    auto zero = [&](float* p, long n) {
        fill_kernel<<<dim3((unsigned)((n + 255) / 256)), dim3(256), 0, stream>>>(p, n);
    };

    // 1) frequency / recency distributions
    stats_kernel<<<dim3(Bc), dim3(256), 0, stream>>>(Xs, decay, fd, rd);

    // 2) embedding gather + concat (padded K=704)
    loc_stage_kernel<<<dim3(Bc * Tc), dim3(256), 0, stream>>>(
        l11, l13, l14, Xs, e11, e13, e14, eX, locbuf);
    pad_copy_kernel<<<dim3((Dm * COMBP + 255) / 256), dim3(256), 0, stream>>>(
        loc_W, locWp, Dm, COMBR, COMBP);

    // 3) freq/rec embeddings: (32,40000) @ (512,40000)^T, K-split 25 with atomics
    zero(fr_cat, (long)Bc * 1024);
    gemm(fd, VX, freq_W, VX, fr_cat,       1024, Bc, Dm, VX, 25, 1);
    gemm(rd, VX, rec_W,  VX, fr_cat + 512, 1024, Bc, Dm, VX, 25, 1);
    bias(fr_cat,       1024, freq_b, Bc, Dm, 1);
    bias(fr_cat + 512, 1024, rec_b,  Bc, Dm, 1);

    // 4) user_context
    gemm(fr_cat, 1024, comb_W, 1024, uc, Dm, Bc, Dm, 1024, 1, 0);
    bias(uc, Dm, comb_b, Bc, Dm, 0);

    // 5) loc projection + positional encoding
    gemm(locbuf, COMBP, locWp, COMBP, xproj, Dm, Bc * Tc, Dm, COMBP, 1, 0);
    xproj_epi_kernel<<<dim3((Bc * Tc * Dm + 255) / 256), dim3(256), 0, stream>>>(
        xproj, loc_b, pe);

    // 6) GRU input projection + recurrence
    gemm(xproj, Dm, gru_Wih, Dm, gi, 768, Bc * Tc, 768, Dm, 1, 0);
    bias(gi, 768, gru_bih, Bc * Tc, 768, 0);
    gru_kernel<<<dim3(1), dim3(256), 0, stream>>>(gi, gru_Whh, gru_bhh, ghbuf, gruh);

    // 7) gp projection + layernorm
    gemm(gruh, GHd, gp_W, GHd, gln, Dm, Bc * Tc, Dm, GHd, 1, 0);
    bias(gln, Dm, gp_b, Bc * Tc, Dm, 0);
    ln_kernel<<<dim3(Bc * Tc), dim3(256), 0, stream>>>(gln, ln_g, ln_b);

    // 8) Q (strided final-state rows), K, V projections
    const float* fs = gln + (long)(Tc - 1) * Dm;   // row stride T*D per batch
    gemm(fs, Tc * Dm, q_W, Dm, Qb, Dm, Bc, Dm, Dm, 1, 0);
    bias(Qb, Dm, q_b, Bc, Dm, 0);
    gemm(gln, Dm, k_W, Dm, Kb, Dm, Bc * Tc, Dm, Dm, 1, 0);
    bias(Kb, Dm, k_b, Bc * Tc, Dm, 0);
    gemm(gln, Dm, v_W, Dm, Vb, Dm, Bc * Tc, Dm, Dm, 1, 0);
    bias(Vb, Dm, v_b, Bc * Tc, Dm, 0);

    // 9) attention + output projection
    attn_kernel<<<dim3(Bc * Hh), dim3(128), 0, stream>>>(Qb, Kb, Vb, attnw, ctxb);
    gemm(ctxb, Dm, o_W, Dm, attd, Dm, Bc, Dm, Dm, 1, 0);
    bias(attd, Dm, o_b, Bc, Dm, 0);

    // 10) combined state -> hidden
    combined_kernel<<<dim3((Bc * Dm + 255) / 256), dim3(256), 0, stream>>>(
        gln, uc, attd, user_emb, usr, comb);
    gemm(comb, Dm, fc_W, Dm, hid, HIDd, Bc, HIDd, Dm, 1, 0);
    bias(hid, HIDd, fc_b, Bc, HIDd, 1);

    // 11) classifier heads
    gemm(hid, HIDd, c11_W, HIDd, out11, V11, Bc, V11, HIDd, 1, 0);
    bias(out11, V11, c11_b, Bc, V11, 0);
    gemm(hid, HIDd, c13_W, HIDd, out13, V13, Bc, V13, HIDd, 1, 0);
    bias(out13, V13, c13_b, Bc, V13, 0);
    gemm(hid, HIDd, c14_W, HIDd, out14, V14, Bc, V14, HIDd, 1, 0);
    bias(out14, V14, c14_b, Bc, V14, 0);
    gemm(hid, HIDd, cX_W, HIDd, gen, VX, Bc, VX, HIDd, 1, 0);
    bias(gen, VX, cX_b, Bc, VX, 0);

    // 12) copy gate
    zero(g1, (long)Bc * Dm);
    gemm(fs,   Tc * Dm, g1_W,        1536, g1, Dm, Bc, Dm, Dm, 1, 1);
    gemm(uc,   Dm,      g1_W + 512,  1536, g1, Dm, Bc, Dm, Dm, 1, 1);
    gemm(attd, Dm,      g1_W + 1024, 1536, g1, Dm, Bc, Dm, Dm, 1, 1);
    bias(g1, Dm, g1_b, Bc, Dm, 1);
    g2_kernel<<<dim3(Bc), dim3(64), 0, stream>>>(g1, g2_W, g2_b, cp);

    // 13) copy distribution + final mixed log-probabilities
    copydist_kernel<<<dim3(Bc), dim3(256), 0, stream>>>(Xs, attnw, fd, rd, cd);
    final_kernel<<<dim3(Bc), dim3(256), 0, stream>>>(gen, cd, cp, outX);
}
